// DiffusionModel_3410204033536
// MI455X (gfx1250) — matrix operand order repaired and checked by reading, not compiled, not run on hardware
//
#include <hip/hip_runtime.h>

#define N_NODES 50000
#define IN_F 64
#define H_F 128
#define S_W 4
#define R_TOT (S_W * N_NODES)

typedef __bf16 bf16;
typedef bf16  v16bf __attribute__((ext_vector_type(16)));
typedef bf16  v2bf  __attribute__((ext_vector_type(2)));
typedef float v8f   __attribute__((ext_vector_type(8)));
typedef float v4f   __attribute__((ext_vector_type(4)));
typedef float v2f   __attribute__((ext_vector_type(2)));
typedef unsigned int v4u __attribute__((ext_vector_type(4)));
typedef int   v4i   __attribute__((ext_vector_type(4)));
typedef int   v8i   __attribute__((ext_vector_type(8)));
typedef unsigned short u16;

union Frag16 { v4u q[2]; v16bf v; };

// Native f32->bf16 (RTNE). Scalar form.
__device__ __forceinline__ u16 f2bf(float f) {
  bf16 b = (bf16)f;
  return __builtin_bit_cast(u16, b);
}
// Packed pair form: one v_cvt_pk_bf16_f32, no v_mov_b16 stitching.
__device__ __forceinline__ unsigned pk2(float lo, float hi) {
  v2f t; t.x = lo; t.y = hi;
  v2bf b = __builtin_convertvector(t, v2bf);
  return __builtin_bit_cast(unsigned, b);
}

// TDM: 2D bf16 tile global->LDS, row padding via D# pad fields.
// padIntervalCode k -> pad every 2^(k+1) dwords; padAmountCode a -> (a+1) dwords pad.
__device__ __forceinline__ void tdm_load_2d_bf16(unsigned ldsByte, const void* gsrc,
                                                 unsigned dim0, unsigned rows,
                                                 unsigned padIntervalCode, unsigned padAmountCode) {
  unsigned long long ga = (unsigned long long)(size_t)gsrc;
  v4u g0;
  g0.x = 1u;                                   // count=1 user descriptor
  g0.y = ldsByte;                              // lds_addr (low 32b of generic LDS ptr)
  g0.z = (unsigned)ga;                         // global_addr[31:0]
  g0.w = (unsigned)(ga >> 32) | 0x80000000u;   // global_addr[56:32] | type=2
  unsigned e0 = (1u << 16)                     // data_size = 2 bytes
              | (1u << 20)                     // pad_enable
              | (padIntervalCode << 22)
              | (padAmountCode   << 25);
  unsigned e1 = (dim0 & 0xFFFFu) << 16;                  // tensor_dim0[15:0]
  unsigned e2 = (dim0 >> 16) | ((rows & 0xFFFFu) << 16); // tensor_dim0 hi | tensor_dim1 lo
  unsigned e3 = (rows >> 16) | ((dim0 & 0xFFFFu) << 16); // tensor_dim1 hi | tile_dim0
  unsigned e4 = (rows & 0xFFFFu);                        // tile_dim1, tile_dim2=0
  unsigned e5 = dim0;                                    // tensor_dim0_stride (elements)
  v8i g1 = { (int)e0, (int)e1, (int)e2, (int)e3, (int)e4, (int)e5, 0, 0 };
  v4i z4 = { 0, 0, 0, 0 };
#if __clang_major__ >= 23
  v8i z8 = { 0, 0, 0, 0, 0, 0, 0, 0 };
  __builtin_amdgcn_tensor_load_to_lds(g0, g1, z4, z4, z8, 0);
#else
  __builtin_amdgcn_tensor_load_to_lds(g0, g1, z4, z4, 0);
#endif
}

__device__ __forceinline__ v16bf lds_frag(const u16* base, unsigned elemOff) {
  Frag16 f;
  const v4u* p = (const v4u*)(base + elemOff);  // 16B aligned (stride multiple of 8 elems)
  f.q[0] = p[0];
  f.q[1] = p[2];
  return f.v;
}

// ---------------------------------------------------------------- prep kernel
// bf16-convert weights into workspace (TDM copies them 1:1), fold cos/sin time
// embedding into bias2, compute sqrt(alpha_bar) scalars from t_idx.
__global__ void __launch_bounds__(256) k_prep(const float* __restrict__ W_in,
                                              const float* __restrict__ W_rec,
                                              const float* __restrict__ W_damp,
                                              const float* __restrict__ b_damp,
                                              const int*   __restrict__ t_idx,
                                              u16* __restrict__ wInB,
                                              u16* __restrict__ wRecB,
                                              u16* __restrict__ wDampB,
                                              float* __restrict__ bias2,
                                              float* __restrict__ scal) {
  const int t  = blockIdx.x * blockDim.x + threadIdx.x;
  const int nt = gridDim.x * blockDim.x;
  for (int i = t; i < H_F * IN_F; i += nt) wInB[i] = f2bf(W_in[i]);
  for (int i = t; i < H_F * H_F; i += nt) {
    const int r = i >> 7, c = i & (H_F - 1);
    wRecB[i]  = f2bf(W_rec[r * 256 + c]);    // first half of (128,256); rest multiplies zeros
    wDampB[i] = f2bf(W_damp[r * 130 + c]);   // first 128 cols of (128,130)
  }
  if (t < H_F) {
    const float tv = (float)(*t_idx) / 100.0f;
    bias2[t] = b_damp[t] + cosf(tv) * W_damp[t * 130 + 128]
                         + sinf(tv) * W_damp[t * 130 + 129];
  }
  if (t == 0) {
    float ab = 1.0f;
    const int ti = *t_idx;
    for (int i = 0; i <= ti; ++i) {
      const float beta = 1e-4f + (0.02f - 1e-4f) * (float)i / 99.0f;
      ab *= (1.0f - beta);
    }
    scal[0] = sqrtf(ab);
    scal[1] = sqrtf(1.0f - ab);
  }
}

// ------------------------------------------------------------- projection GEMM
// h1 = bf16( hraw @ W_in^T + b_in ), one 16x128 output tile per wave.
__global__ void __launch_bounds__(256) k_proj(const float* __restrict__ hraw,
                                              const float* __restrict__ b_in,
                                              const u16* __restrict__ wInB,
                                              u16* __restrict__ h1) {
  extern __shared__ char smem[];
  u16* wlds = (u16*)smem;                        // 128 rows x stride 72 bf16
  const int tid = threadIdx.x;
  const int wave = tid >> 5, lane = tid & 31;
  const int half = lane >> 4, ln = lane & 15;
  if (tid == 0) {
    tdm_load_2d_bf16((unsigned)(size_t)smem, wInB, 64u, 128u, 4u, 3u); // stride 72
    __builtin_amdgcn_s_wait_tensorcnt(0);
  }
  __syncthreads();

  const int rowBase = blockIdx.x * 128 + wave * 16;
  int row = rowBase + ln;
  if (row >= N_NODES) row = N_NODES - 1;

  Frag16 A[2];
#pragma unroll
  for (int kc = 0; kc < 2; ++kc) {
    const v4f* p0 = (const v4f*)(hraw + row * IN_F + kc * 32 + half * 8);
    const v4f* p1 = (const v4f*)(hraw + row * IN_F + kc * 32 + 16 + half * 8);
    v4f f0 = p0[0], f1 = p0[1], f2 = p1[0], f3 = p1[1];
    v4u q;
    q.x = pk2(f0.x, f0.y); q.y = pk2(f0.z, f0.w);
    q.z = pk2(f1.x, f1.y); q.w = pk2(f1.z, f1.w);
    A[kc].q[0] = q;
    q.x = pk2(f2.x, f2.y); q.y = pk2(f2.z, f2.w);
    q.z = pk2(f3.x, f3.y); q.w = pk2(f3.z, f3.w);
    A[kc].q[1] = q;
  }

  v8f acc[8] = {};
#pragma unroll
  for (int kc = 0; kc < 2; ++kc)
#pragma unroll
    for (int j = 0; j < 8; ++j) {
      v16bf B = lds_frag(wlds, (unsigned)((j * 16 + ln) * 72 + kc * 32 + half * 8));
      acc[j] = __builtin_amdgcn_wmma_f32_16x16x32_bf16(false, A[kc].v, false, B,
                                                       (short)0, acc[j], false, false);
    }

#pragma unroll
  for (int j = 0; j < 8; ++j) {
    const int col = j * 16 + ln;
    const float bi = b_in[col];
#pragma unroll
    for (int v = 0; v < 8; ++v) {
      const int m = rowBase + half * 8 + v;
      if (m < N_NODES) h1[m * H_F + col] = f2bf(acc[j][v] + bi);
    }
  }
}

// -------------------------------------------------------------- fused main op
// Per 16 (s,n) rows per wave: gather h1[walks[..,0]], z=tanh(.@Wrec^T+b_rec),
// damping=z@Wdamp^T+bias2, row-reduce, diffusion loss accumulate.
__global__ void __launch_bounds__(256) k_main(const u16* __restrict__ h1,
    const int* __restrict__ walks, const float* __restrict__ x,
    const float* __restrict__ eps, const float* __restrict__ b_rec,
    const float* __restrict__ bias2, const float* __restrict__ b_proj,
    const u16* __restrict__ wRecB, const u16* __restrict__ wDampB,
    const float* __restrict__ scal, float* __restrict__ accum) {
  extern __shared__ char smem[];
  u16*   wrec = (u16*)smem;                 // 128 x 136 bf16  (34816 B)
  u16*   wdmp = (u16*)(smem + 34816);       // 128 x 136 bf16  (34816 B)
  u16*   zall = (u16*)(smem + 69632);       // 8 waves x 16 x 136 (34816 B)
  float* red  = (float*)(smem + 104448);    // 8 waves x 16 x {sd,sb}
  const int tid = threadIdx.x;
  const int wave = tid >> 5, lane = tid & 31;
  const int half = lane >> 4, ln = lane & 15;
  if (tid == 0) {
    tdm_load_2d_bf16((unsigned)(size_t)smem,           wRecB,  128u, 128u, 5u, 3u);
    tdm_load_2d_bf16((unsigned)(size_t)(smem + 34816), wDampB, 128u, 128u, 5u, 3u);
    __builtin_amdgcn_s_wait_tensorcnt(0);
  }
  __syncthreads();

  const int rowBase = blockIdx.x * 128 + wave * 16;
  int g = rowBase + ln;
  if (g >= R_TOT) g = R_TOT - 1;
  const int idx0 = walks[g * 2 + 0];          // walk step 0 gather index

  u16* zw = zall + wave * (16 * 136);

  // ---- matmul 1: tanh(h1[idx0] @ Wrec[:, :128]^T + b_rec)
  v8f acc1[8] = {};
#pragma unroll
  for (int kc = 0; kc < 4; ++kc) {
    Frag16 A;
    A.q[0] = *(const v4u*)(h1 + idx0 * H_F + kc * 32 + half * 8);
    A.q[1] = *(const v4u*)(h1 + idx0 * H_F + kc * 32 + 16 + half * 8);
#pragma unroll
    for (int j = 0; j < 8; ++j) {
      v16bf B = lds_frag(wrec, (unsigned)((j * 16 + ln) * 136 + kc * 32 + half * 8));
      acc1[j] = __builtin_amdgcn_wmma_f32_16x16x32_bf16(false, A.v, false, B,
                                                        (short)0, acc1[j], false, false);
    }
  }
  // C-layout -> LDS (bf16, A-layout-friendly row-major, stride 136)
#pragma unroll
  for (int j = 0; j < 8; ++j) {
    const int col = j * 16 + ln;
    const float br = b_rec[col];
#pragma unroll
    for (int v = 0; v < 8; ++v)
      zw[(half * 8 + v) * 136 + col] = f2bf(tanhf(acc1[j][v] + br));
  }

  // ---- matmul 2: damping = z @ Wdamp[:, :128]^T (+ bias2 later)
  v8f acc2[8] = {};
#pragma unroll
  for (int kc = 0; kc < 4; ++kc) {
    Frag16 A;
    A.q[0] = *(const v4u*)(zw + ln * 136 + kc * 32 + half * 8);
    A.q[1] = *(const v4u*)(zw + ln * 136 + kc * 32 + 16 + half * 8);
#pragma unroll
    for (int j = 0; j < 8; ++j) {
      v16bf B = lds_frag(wdmp, (unsigned)((j * 16 + ln) * 136 + kc * 32 + half * 8));
      acc2[j] = __builtin_amdgcn_wmma_f32_16x16x32_bf16(false, A.v, false, B,
                                                        (short)0, acc2[j], false, false);
    }
  }

  // ---- row reductions: sd = sum_k damping, sb = sum_k b_proj[k]*damping
  float ps[8], pb[8];
#pragma unroll
  for (int v = 0; v < 8; ++v) { ps[v] = 0.f; pb[v] = 0.f; }
#pragma unroll
  for (int j = 0; j < 8; ++j) {
    const int col = j * 16 + ln;
    const float b2 = bias2[col], bp = b_proj[col];
#pragma unroll
    for (int v = 0; v < 8; ++v) {
      const float d = acc2[j][v] + b2;
      ps[v] += d;
      pb[v] += d * bp;
    }
  }
#pragma unroll
  for (int m = 1; m < 16; m <<= 1) {
#pragma unroll
    for (int v = 0; v < 8; ++v) {
      ps[v] += __shfl_xor(ps[v], m, 32);
      pb[v] += __shfl_xor(pb[v], m, 32);
    }
  }
  if (ln == 0) {
#pragma unroll
    for (int v = 0; v < 8; ++v) {
      red[(wave * 16 + half * 8 + v) * 2 + 0] = ps[v];
      red[(wave * 16 + half * 8 + v) * 2 + 1] = pb[v];
    }
  }
  // wave-synchronous LDS round trip (writer lanes 0 & 16, readers 0..15)
  float lsum = 0.f;
  if (lane < 16) {
    const int gr = rowBase + lane;
    if (gr < R_TOT) {
      const float sd = red[(wave * 16 + lane) * 2 + 0];
      const float sb = red[(wave * 16 + lane) * 2 + 1];
      const int i1 = walks[gr * 2 + 1];
      const float sa = scal[0], se = scal[1];
#pragma unroll
      for (int c = 0; c < 3; ++c) {
        const float ev = eps[gr * 3 + c];
        const float xt = sa * x[i1 * 3 + c] + se * ev;
        const float pred = (xt * sd + sb) * (1.0f / (float)H_F);
        const float d = pred - ev;
        lsum += d * d;
      }
    }
  }
#pragma unroll
  for (int m = 1; m < 32; m <<= 1) lsum += __shfl_xor(lsum, m, 32);
  if (lane == 0) atomicAdd(accum, lsum);
}

__global__ void k_fin(const float* __restrict__ accum, float* __restrict__ out) {
  out[0] = accum[0] * (1.0f / (float)(R_TOT * 3));
}

extern "C" void kernel_launch(void* const* d_in, const int* in_sizes, int n_in,
                              void* d_out, int out_size, void* d_ws, size_t ws_size,
                              hipStream_t stream) {
  const float* hraw   = (const float*)d_in[0];
  const float* x      = (const float*)d_in[1];
  const int*   walks  = (const int*)  d_in[2];
  const float* eps    = (const float*)d_in[3];
  const int*   t_idx  = (const int*)  d_in[4];
  const float* W_in   = (const float*)d_in[5];
  const float* b_in   = (const float*)d_in[6];
  const float* W_rec  = (const float*)d_in[7];
  const float* b_rec  = (const float*)d_in[8];
  // d_in[9..12] (W_gate,b_gate,W_mix,b_mix) and d_in[15] (W_proj) are dead.
  const float* W_damp = (const float*)d_in[13];
  const float* b_damp = (const float*)d_in[14];
  const float* b_proj = (const float*)d_in[16];

  char* ws = (char*)d_ws;
  float* scal   = (float*)(ws + 0);
  float* bias2  = (float*)(ws + 256);
  float* accum  = (float*)(ws + 768);
  u16*   wInB   = (u16*)(ws + 1024);                      // 16384 B
  u16*   wRecB  = (u16*)(ws + 1024 + 16384);              // 32768 B
  u16*   wDampB = (u16*)(ws + 1024 + 16384 + 32768);      // 32768 B
  u16*   h1     = (u16*)(ws + 1024 + 16384 + 65536);      // N*128*2 B (bf16, L2-resident)

  hipMemsetAsync(accum, 0, 256, stream);
  k_prep<<<64, 256, 0, stream>>>(W_in, W_rec, W_damp, b_damp, t_idx,
                                 wInB, wRecB, wDampB, bias2, scal);

  const int blocksProj = (N_NODES + 127) / 128;   // 391
  k_proj<<<blocksProj, 256, 128 * 72 * 2, stream>>>(hraw, b_in, wInB, h1);

  const int blocksMain = (R_TOT + 127) / 128;     // 1563
  const size_t ldsMain = 34816u * 3u + 8u * 16u * 2u * 4u; // 105472 B
  k_main<<<blocksMain, 256, ldsMain, stream>>>(h1, walks, x, eps, b_rec, bias2,
                                               b_proj, wRecB, wDampB, scal, accum);

  k_fin<<<1, 1, 0, stream>>>(accum, (float*)d_out);
}
